// HIIFL_22273700397441
// MI455X (gfx1250) — matrix operand order repaired and checked by reading, not compiled, not run on hardware
//
#include <hip/hip_runtime.h>
#include <hip/hip_bf16.h>

// ---------- types ----------
typedef __attribute__((ext_vector_type(16))) __bf16 v16bf;
typedef __attribute__((ext_vector_type(8)))  __bf16 v8bf;
typedef __attribute__((ext_vector_type(8)))  float  v8f;

// ---------- problem constants ----------
#define BATCH 8
#define CDIM 96
#define HLR 96
#define WLR 96
#define HOUT 192
#define WOUT 192
#define NTOK (HOUT*WOUT)          // 36864 per batch
#define HID 384
#define NH 8
#define HD 12
#define C2 192
#define QKVD 288
#define BTOK 32                   // tokens per block (2 M-subtiles of 16)
#define BLOCKS_PER_B (NTOK/BTOK)  // 1152

// ---------- workspace layout (bytes), offsets multiples of 256 ----------
#define OFF_X     ((size_t)0)                          // x after LN1, f32 [B*N,96]
#define OFF_Q     (OFF_X + (size_t)113246208)          // q softmaxed, f32 [B*N,96]
#define OFF_Z     (OFF_Q + (size_t)113246208)          // Z[b,96] f32
#define OFF_CTX   (OFF_Z + (size_t)3072)               // ctx raw [b,8,12,12] f32
#define OFF_CTXN  (OFF_CTX + (size_t)36864)            // ctx normalized
#define OFF_W1C   (OFF_CTXN + (size_t)36864)           // mlp1_w1 colmajor bf16 [384][192]
#define OFF_W2C   (OFF_W1C + (size_t)147456)           // mlp1_w2 colmajor bf16 [96][384]
#define OFF_QKVC  (OFF_W2C + (size_t)73728)            // qkv_w colmajor bf16 [288][96]
#define OFF_PROJC (OFF_QKVC + (size_t)55296)           // proj_w colmajor bf16 [96][96]
#define OFF_M2W1C (OFF_PROJC + (size_t)18432)          // mlp2_w1 colmajor bf16 [384][96]

// ---------- helpers ----------
static __device__ __forceinline__ float gelu_exact(float x) {
    return 0.5f * x * (1.0f + erff(x * 0.70710678118654752f));
}

// A fragment (16x32 bf16, M x K) from a row-major row pointer in LDS.
// lane l(0-15): row M=l, elements e<8 -> K=k0+e+8h ; e>=8 -> K=k0+e+8+8h
static __device__ __forceinline__ v16bf load_a_frag(const __bf16* rowp, int k0, int h) {
    v8bf lo = *(const v8bf*)(rowp + k0 + 8 * h);
    v8bf hi = *(const v8bf*)(rowp + k0 + 16 + 8 * h);
    v16bf a;
#pragma unroll
    for (int i = 0; i < 8; ++i) { a[i] = lo[i]; a[i + 8] = hi[i]; }
    return a;
}

// B fragment (32x16 bf16, K x N) from col-major weights: per lane column = col,
// elements e<8 -> K = k0 + e + 8h ; e>=8 -> K = k0 + e + 8 + 8h
static __device__ __forceinline__ v16bf load_b_frag(const __bf16* colp, int k0, int h) {
    v8bf lo = *(const v8bf*)(colp + k0 + 8 * h);
    v8bf hi = *(const v8bf*)(colp + k0 + 16 + 8 * h);
    v16bf bb;
#pragma unroll
    for (int i = 0; i < 8; ++i) { bb[i] = lo[i]; bb[i + 8] = hi[i]; }
    return bb;
}

static __device__ __forceinline__ v8f wmma_bf16(v16bf a, v16bf b, v8f c) {
    return __builtin_amdgcn_wmma_f32_16x16x32_bf16(false, a, false, b, (short)0, c, false, false);
}

// ---------- kernel 0: weight conversion to col-major bf16 ----------
__global__ __launch_bounds__(256) void hiifl_prep(
    const float* __restrict__ w1, const float* __restrict__ w2,
    const float* __restrict__ qkv, const float* __restrict__ proj,
    const float* __restrict__ m2w1,
    __bf16* __restrict__ w1c, __bf16* __restrict__ w2c,
    __bf16* __restrict__ qkvc, __bf16* __restrict__ projc,
    __bf16* __restrict__ m2w1c)
{
    int i = blockIdx.x * 256 + threadIdx.x;
    if (i < 73728) { int k = i / 384, n = i % 384; w1c[n * 192 + k] = (__bf16)w1[i]; return; }
    i -= 73728;
    if (i < 36864) { int k = i / 96, n = i % 96; w2c[n * 384 + k] = (__bf16)w2[i]; return; }
    i -= 36864;
    if (i < 27648) { int k = i / 288, n = i % 288; qkvc[n * 96 + k] = (__bf16)qkv[i]; return; }
    i -= 27648;
    if (i < 9216)  { int k = i / 96, n = i % 96; projc[n * 96 + k] = (__bf16)proj[i]; return; }
    i -= 9216;
    if (i < 36864) { int k = i / 384, n = i % 384; m2w1c[n * 96 + k] = (__bf16)m2w1[i]; return; }
}

// ---------- kernel 1: upsample + coord-enc + MLP1 + LN1 + QKV + attention reductions ----------
// LDS arena (lifetime-overlaid), 43 KB:
//   [0,     12288) comb bf16[32][192]   (stage1 A)   -> xf f32[32][96] (stage2 out / LN / exp(k))
//   [12288, 36864) hidb bf16[32][384]   (stage1->2)  -> qkvb bf16[32][288] (stage3 out)
//   [36864, 43008) xb bf16[32][96]      (LN out, stage3 A)
__global__ __launch_bounds__(256) void hiifl_front(
    const float* __restrict__ feat,
    const __bf16* __restrict__ w1c, const float* __restrict__ b1,
    const __bf16* __restrict__ w2c, const float* __restrict__ b2,
    const float* __restrict__ n1g, const float* __restrict__ n1b,
    const __bf16* __restrict__ qkvc,
    float* __restrict__ x_ws, float* __restrict__ q_ws,
    float* __restrict__ Zb, float* __restrict__ ctxb)
{
    __shared__ __align__(128) unsigned char smem[43008];
    __shared__ float mu_s[BTOK], rs_s[BTOK];
    __bf16 (*comb)[C2]   = (__bf16(*)[C2])  (smem + 0);
    float  (*xf)[CDIM]   = (float(*)[CDIM]) (smem + 0);
    __bf16 (*hidb)[HID]  = (__bf16(*)[HID]) (smem + 12288);
    __bf16 (*qkvb)[QKVD] = (__bf16(*)[QKVD])(smem + 12288);
    __bf16 (*xb)[CDIM]   = (__bf16(*)[CDIM])(smem + 36864);

    const int tid = threadIdx.x;
    const int b   = blockIdx.x / BLOCKS_PER_B;
    const int t0  = (blockIdx.x % BLOCKS_PER_B) * BTOK;
    const int wv = tid >> 5, lane = tid & 31, r = lane & 15, h = lane >> 4;

    // ---- build combined = [bilinear feat | coord enc] in bf16 ----
    for (int idx = tid; idx < BTOK * C2; idx += 256) {
        int tt = idx / C2, cc = idx % C2;
        int n = t0 + tt, py = n / WOUT, px = n % WOUT;
        float val;
        if (cc < CDIM) {
            float sy = fminf(fmaxf((py + 0.5f) * 0.5f - 0.5f, 0.0f), 95.0f);
            float sx = fminf(fmaxf((px + 0.5f) * 0.5f - 0.5f, 0.0f), 95.0f);
            int y0 = (int)sy, x0 = (int)sx;
            int y1 = min(y0 + 1, 95), x1 = min(x0 + 1, 95);
            float wy = sy - (float)y0, wx = sx - (float)x0;
            const float* fp = feat + (((size_t)(b * CDIM + cc)) * HLR) * WLR;
            float v00 = fp[y0 * WLR + x0], v01 = fp[y0 * WLR + x1];
            float v10 = fp[y1 * WLR + x0], v11 = fp[y1 * WLR + x1];
            val = v00 * (1.f - wy) * (1.f - wx) + v01 * (1.f - wy) * wx
                + v10 * wy * (1.f - wx) + v11 * wy * wx;
        } else {
            int ch = cc - CDIM;
            int f = ch >> 3, rem = ch & 7, p = rem >> 1, s = rem & 1;
            float yn = (float)py * (1.0f / 191.0f), xn = (float)px * (1.0f / 191.0f);
            float c4 = (p == 0) ? yn : (p == 1) ? xn : (p == 2) ? yn : xn;  // cs = 2/scale = 1
            float ph = ldexpf(c4 * 3.14159265358979323846f, f);
            val = s ? cosf(ph) : sinf(ph);
        }
        comb[tt][cc] = (__bf16)val;
    }
    __syncthreads();

    // ---- MLP1 stage 1: [32,192] @ [192,384] -> GELU -> hidb (bf16) ----
    for (int ct = wv; ct < 24; ct += 8) {
        v8f acc0 = {}, acc1 = {};
        const int col = ct * 16 + r;
        const __bf16* bp = w1c + (size_t)col * 192;
        const __bf16* a0 = &comb[r][0];
        const __bf16* a1 = &comb[16 + r][0];
#pragma unroll
        for (int kk = 0; kk < 192; kk += 32) {
            v16bf bw = load_b_frag(bp, kk, h);
            acc0 = wmma_bf16(load_a_frag(a0, kk, h), bw, acc0);
            acc1 = wmma_bf16(load_a_frag(a1, kk, h), bw, acc1);
        }
        float bias = b1[col];
#pragma unroll
        for (int v = 0; v < 8; ++v) {
            hidb[v + 8 * h][col]      = (__bf16)gelu_exact(acc0[v] + bias);
            hidb[16 + v + 8 * h][col] = (__bf16)gelu_exact(acc1[v] + bias);
        }
    }
    __syncthreads();

    // ---- MLP1 stage 2: [32,384] @ [384,96] + b2 -> xf (f32, aliases comb) ----
    if (wv < 6) {
        v8f acc0 = {}, acc1 = {};
        const int col = wv * 16 + r;
        const __bf16* bp = w2c + (size_t)col * 384;
        const __bf16* a0 = &hidb[r][0];
        const __bf16* a1 = &hidb[16 + r][0];
#pragma unroll
        for (int kk = 0; kk < 384; kk += 32) {
            v16bf bw = load_b_frag(bp, kk, h);
            acc0 = wmma_bf16(load_a_frag(a0, kk, h), bw, acc0);
            acc1 = wmma_bf16(load_a_frag(a1, kk, h), bw, acc1);
        }
        float bias = b2[col];
#pragma unroll
        for (int v = 0; v < 8; ++v) {
            xf[v + 8 * h][col]      = acc0[v] + bias;
            xf[16 + v + 8 * h][col] = acc1[v] + bias;
        }
    }
    __syncthreads();

    // ---- LayerNorm1 ----
    if (tid < BTOK) {
        float s = 0.f, s2 = 0.f;
        for (int c = 0; c < CDIM; ++c) { float v = xf[tid][c]; s += v; s2 += v * v; }
        float mu = s * (1.0f / 96.0f);
        float var = s2 * (1.0f / 96.0f) - mu * mu;
        mu_s[tid] = mu; rs_s[tid] = rsqrtf(var + 1e-5f);
    }
    __syncthreads();
    for (int idx = tid; idx < BTOK * CDIM; idx += 256) {
        int tt = idx / CDIM, c = idx % CDIM;
        float v = (xf[tt][c] - mu_s[tt]) * rs_s[tt] * n1g[c] + n1b[c];
        xb[tt][c] = (__bf16)v;
        x_ws[((size_t)(b * NTOK + t0 + tt)) * CDIM + c] = v;
    }
    __syncthreads();

    // ---- QKV: [32,96] @ [96,288] -> qkvb (bf16, aliases hidb) ----
    for (int ct = wv; ct < 18; ct += 8) {
        v8f acc0 = {}, acc1 = {};
        const int col = ct * 16 + r;
        const __bf16* bp = qkvc + (size_t)col * 96;
        const __bf16* a0 = &xb[r][0];
        const __bf16* a1 = &xb[16 + r][0];
#pragma unroll
        for (int kk = 0; kk < 96; kk += 32) {
            v16bf bw = load_b_frag(bp, kk, h);
            acc0 = wmma_bf16(load_a_frag(a0, kk, h), bw, acc0);
            acc1 = wmma_bf16(load_a_frag(a1, kk, h), bw, acc1);
        }
#pragma unroll
        for (int v = 0; v < 8; ++v) {
            qkvb[v + 8 * h][col]      = (__bf16)acc0[v];
            qkvb[16 + v + 8 * h][col] = (__bf16)acc1[v];
        }
    }
    __syncthreads();

    // ---- q: softmax over head-dim (12), * hd^-0.5 -> q_ws (256 tasks = 256 threads) ----
    {
        int tt = tid >> 3, hh = tid & 7;
        float qv[HD];
        float m = -1e30f;
        for (int d = 0; d < HD; ++d) { qv[d] = (float)qkvb[tt][hh * HD + d]; m = fmaxf(m, qv[d]); }
        float ssum = 0.f;
        for (int d = 0; d < HD; ++d) { qv[d] = expf(qv[d] - m); ssum += qv[d]; }
        float inv = rsqrtf(12.0f) / ssum;
        for (int d = 0; d < HD; ++d)
            q_ws[((size_t)(b * NTOK + t0 + tt)) * CDIM + hh * HD + d] = qv[d] * inv;
    }
    // ---- exp(k) into xf (aliases comb/stage2-out; both dead) ----
    for (int idx = tid; idx < BTOK * CDIM; idx += 256) {
        int tt = idx / CDIM, d = idx % CDIM;
        xf[tt][d] = expf((float)qkvb[tt][CDIM + d]);
    }
    __syncthreads();

    // ---- Z partials ----
    if (tid < CDIM) {
        float s = 0.f;
        for (int tt = 0; tt < BTOK; ++tt) s += xf[tt][tid];
        atomicAdd(&Zb[b * CDIM + tid], s);
    }
    // ---- ctx partials: [8,12,12] per batch ----
    for (int idx = tid; idx < NH * HD * HD; idx += 256) {
        int hh = idx / 144, rr = idx % 144, d = rr / HD, e = rr % HD;
        float s = 0.f;
        for (int tt = 0; tt < BTOK; ++tt)
            s += xf[tt][hh * HD + d] * (float)qkvb[tt][2 * CDIM + hh * HD + e];
        atomicAdd(&ctxb[(size_t)b * 1152 + idx], s);
    }
}

// ---------- kernel 2: normalize ctx by Z ----------
__global__ __launch_bounds__(256) void hiifl_ctx(
    const float* __restrict__ ctxb, const float* __restrict__ Zb,
    float* __restrict__ ctxn)
{
    int i = blockIdx.x * 256 + threadIdx.x;
    if (i < BATCH * NH * HD * HD) {
        int b = i / 1152, rem = i % 1152, hh = rem / 144, d = (rem % 144) / HD;
        ctxn[i] = ctxb[i] / Zb[b * CDIM + hh * HD + d];
    }
}

// ---------- kernel 3: attn + proj + LN2 + MLP2 + output ----------
// LDS arena (lifetime-overlaid), 52.5 KB:
//   [0,     12288) xf f32[32][96]   (residual; dead after proj)   \  hidb bf16[32][384]
//   [12288, 24576) qs f32[32][96]   (dead after attn)             /  (MLP2 s1 out)
//   [24576, 29184) ctxs f32[1152]   (dead after attn)
//   [29184, 35328) ab bf16[32][96]  (attn out / proj A)
//   [35328, 47616) yf f32[32][96]   (proj out / LN2 in)
//   [47616, 53760) yb bf16[32][96]  (LN2 out / MLP2 A)
__global__ __launch_bounds__(256) void hiifl_back(
    const float* __restrict__ x_ws, const float* __restrict__ q_ws,
    const float* __restrict__ ctxn,
    const __bf16* __restrict__ projc, const float* __restrict__ projb,
    const float* __restrict__ n2g, const float* __restrict__ n2b,
    const __bf16* __restrict__ m2w1c, const float* __restrict__ m2b1,
    const float* __restrict__ m2w2, const float* __restrict__ m2b2,
    float* __restrict__ out)
{
    __shared__ __align__(128) unsigned char smem[53760];
    __shared__ float mu_s[BTOK], rs_s[BTOK];
    float  (*xf)[CDIM]  = (float(*)[CDIM]) (smem + 0);
    float  (*qs)[CDIM]  = (float(*)[CDIM]) (smem + 12288);
    float  *ctxs        = (float*)         (smem + 24576);
    __bf16 (*ab)[CDIM]  = (__bf16(*)[CDIM])(smem + 29184);
    float  (*yf)[CDIM]  = (float(*)[CDIM]) (smem + 35328);
    __bf16 (*yb)[CDIM]  = (__bf16(*)[CDIM])(smem + 47616);
    __bf16 (*hidb)[HID] = (__bf16(*)[HID]) (smem + 0);       // aliases xf+qs

    const int tid = threadIdx.x;
    const int b   = blockIdx.x / BLOCKS_PER_B;
    const int t0  = (blockIdx.x % BLOCKS_PER_B) * BTOK;
    const int wv = tid >> 5, lane = tid & 31, r = lane & 15, h = lane >> 4;

    for (int idx = tid; idx < BTOK * CDIM; idx += 256) {
        int tt = idx / CDIM, c = idx % CDIM;
        size_t g = ((size_t)(b * NTOK + t0 + tt)) * CDIM + c;
        xf[tt][c] = x_ws[g];
        qs[tt][c] = q_ws[g];
    }
    for (int idx = tid; idx < NH * HD * HD; idx += 256)
        ctxs[idx] = ctxn[(size_t)b * 1152 + idx];
    __syncthreads();

    // ---- attn = q @ ctx  (per head 12x12) ----
    for (int idx = tid; idx < BTOK * CDIM; idx += 256) {
        int tt = idx / CDIM, col = idx % CDIM;
        int hh = col / HD, e = col % HD;
        float s = 0.f;
#pragma unroll
        for (int d = 0; d < HD; ++d)
            s += qs[tt][hh * HD + d] * ctxs[hh * 144 + d * HD + e];
        ab[tt][col] = (__bf16)s;
    }
    __syncthreads();

    // ---- proj: [32,96] @ [96,96] + b + residual -> yf ----
    if (wv < 6) {
        v8f acc0 = {}, acc1 = {};
        const int col = wv * 16 + r;
        const __bf16* bp = projc + (size_t)col * 96;
        const __bf16* a0 = &ab[r][0];
        const __bf16* a1 = &ab[16 + r][0];
#pragma unroll
        for (int kk = 0; kk < 96; kk += 32) {
            v16bf bw = load_b_frag(bp, kk, h);
            acc0 = wmma_bf16(load_a_frag(a0, kk, h), bw, acc0);
            acc1 = wmma_bf16(load_a_frag(a1, kk, h), bw, acc1);
        }
        float bias = projb[col];
#pragma unroll
        for (int v = 0; v < 8; ++v) {
            int r0 = v + 8 * h, r1 = 16 + v + 8 * h;
            yf[r0][col] = acc0[v] + bias + xf[r0][col];
            yf[r1][col] = acc1[v] + bias + xf[r1][col];
        }
    }
    __syncthreads();

    // ---- LayerNorm2 -> yb (bf16) ----
    if (tid < BTOK) {
        float s = 0.f, s2 = 0.f;
        for (int c = 0; c < CDIM; ++c) { float v = yf[tid][c]; s += v; s2 += v * v; }
        float mu = s * (1.0f / 96.0f);
        float var = s2 * (1.0f / 96.0f) - mu * mu;
        mu_s[tid] = mu; rs_s[tid] = rsqrtf(var + 1e-5f);
    }
    __syncthreads();
    for (int idx = tid; idx < BTOK * CDIM; idx += 256) {
        int tt = idx / CDIM, c = idx % CDIM;
        yb[tt][c] = (__bf16)((yf[tt][c] - mu_s[tt]) * rs_s[tt] * n2g[c] + n2b[c]);
    }
    __syncthreads();

    // ---- MLP2 stage 1: [32,96] @ [96,384] -> GELU -> hidb (aliases xf+qs) ----
    for (int ct = wv; ct < 24; ct += 8) {
        v8f acc0 = {}, acc1 = {};
        const int col = ct * 16 + r;
        const __bf16* bp = m2w1c + (size_t)col * 96;
        const __bf16* a0 = &yb[r][0];
        const __bf16* a1 = &yb[16 + r][0];
#pragma unroll
        for (int kk = 0; kk < 96; kk += 32) {
            v16bf bw = load_b_frag(bp, kk, h);
            acc0 = wmma_bf16(load_a_frag(a0, kk, h), bw, acc0);
            acc1 = wmma_bf16(load_a_frag(a1, kk, h), bw, acc1);
        }
        float bias = m2b1[col];
#pragma unroll
        for (int v = 0; v < 8; ++v) {
            hidb[v + 8 * h][col]      = (__bf16)gelu_exact(acc0[v] + bias);
            hidb[16 + v + 8 * h][col] = (__bf16)gelu_exact(acc1[v] + bias);
        }
    }
    __syncthreads();

    // ---- MLP2 stage 2 (384 -> 3) + tanh/clip + NCHW scatter ----
    if (tid < BTOK * 3) {
        int tt = tid / 3, c = tid % 3;
        float s = m2b2[c];
        for (int j = 0; j < HID; ++j)
            s += (float)hidb[tt][j] * m2w2[j * 3 + c];
        float rv = tanhf(s) * 0.5f + 0.5f;
        rv = fminf(fmaxf(rv, 0.0f), 1.0f);
        int n = t0 + tt, py = n / WOUT, px = n % WOUT;
        out[((size_t)(b * 3 + c) * HOUT + py) * WOUT + px] = rv;
    }
}

extern "C" void kernel_launch(void* const* d_in, const int* in_sizes, int n_in,
                              void* d_out, int out_size, void* d_ws, size_t ws_size,
                              hipStream_t stream) {
    const float* feat  = (const float*)d_in[0];
    // d_in[1] = scale (==2, baked into constants)
    const float* w1    = (const float*)d_in[2];
    const float* b1    = (const float*)d_in[3];
    const float* w2    = (const float*)d_in[4];
    const float* b2    = (const float*)d_in[5];
    const float* n1g   = (const float*)d_in[6];
    const float* n1b   = (const float*)d_in[7];
    const float* qkvw  = (const float*)d_in[8];
    const float* projw = (const float*)d_in[9];
    const float* projb = (const float*)d_in[10];
    const float* n2g   = (const float*)d_in[11];
    const float* n2b   = (const float*)d_in[12];
    const float* m2w1  = (const float*)d_in[13];
    const float* m2b1  = (const float*)d_in[14];
    const float* m2w2  = (const float*)d_in[15];
    const float* m2b2  = (const float*)d_in[16];

    char* ws = (char*)d_ws;
    float*  x_ws   = (float*)(ws + OFF_X);
    float*  q_ws   = (float*)(ws + OFF_Q);
    float*  Zb     = (float*)(ws + OFF_Z);
    float*  ctxb   = (float*)(ws + OFF_CTX);
    float*  ctxn   = (float*)(ws + OFF_CTXN);
    __bf16* w1c    = (__bf16*)(ws + OFF_W1C);
    __bf16* w2c    = (__bf16*)(ws + OFF_W2C);
    __bf16* qkvc   = (__bf16*)(ws + OFF_QKVC);
    __bf16* projc  = (__bf16*)(ws + OFF_PROJC);
    __bf16* m2w1c  = (__bf16*)(ws + OFF_M2W1C);
    float*  outp   = (float*)d_out;

    // zero Z + ctx accumulators (contiguous region)
    hipMemsetAsync(ws + OFF_Z, 0, 3072 + 36864, stream);

    hiifl_prep<<<720, 256, 0, stream>>>(w1, w2, qkvw, projw, m2w1,
                                        w1c, w2c, qkvc, projc, m2w1c);

    const int nblk = BATCH * BLOCKS_PER_B;  // 9216
    hiifl_front<<<nblk, 256, 0, stream>>>(feat, w1c, b1, w2c, b2, n1g, n1b,
                                          qkvc, x_ws, q_ws, Zb, ctxb);

    hiifl_ctx<<<36, 256, 0, stream>>>(ctxb, Zb, ctxn);

    hiifl_back<<<nblk, 256, 0, stream>>>(x_ws, q_ws, ctxn, projc, projb,
                                         n2g, n2b, m2w1c, m2b1, m2w2, m2b2, outp);
}
